// AttentiveTransformer_63436666962468
// MI455X (gfx1250) — compile-verified, matrix operand order checked by reading, not run on hardware
//
#include <hip/hip_runtime.h>
#include <hip/hip_bf16.h>

#define D_DIM  512
#define BATCHN 131072
#define BN_EPS 1e-5f

typedef __bf16 bf16_t;
typedef bf16_t v16bf __attribute__((ext_vector_type(16)));
typedef float  v8f   __attribute__((ext_vector_type(8)));

// ---------------------------------------------------------------------------
// Kernel 1: fused  (priors*feat -> BN eval -> GEMM vs W^T -> +bias)
//
// Workgroup: 256 threads = 8 waves (4 along M x 2 along N)
// WG tile: 128(M) x 128(N); wave tile: 32(M) x 64(N) = 2x4 WMMA tiles.
// K loop over 512 in steps of 32 with v_wmma_f32_16x16x32_bf16, 3-term bf16
// split (aHi*bHi + aLo*bHi + aHi*bLo) for ~fp32 accuracy.
//
// Per K-step, the 8 waves cooperatively convert the A block (128x32, BN
// fused) and B block (128 cols x 32 K of W) into hi/lo bf16 planes in LDS,
// stored in WMMA-fragment order (tile x lane x 16 elems). This removes the
// 4x (B) / 2x (A) per-wave conversion redundancy so the f32->bf16x2 split
// VALU work co-executes cleanly under the WMMA stream.
// ---------------------------------------------------------------------------
__global__ __launch_bounds__(256)
void at_gemm_bn_kernel(const float* __restrict__ priors,
                       const float* __restrict__ feat,
                       const float* __restrict__ gamma,
                       const float* __restrict__ beta,
                       const float* __restrict__ mean,
                       const float* __restrict__ var,
                       const float* __restrict__ W,     // [512 out, 512 in] row-major
                       const float* __restrict__ bias,
                       float* __restrict__ out)
{
    __shared__ float s_scale[D_DIM];
    __shared__ float s_shift[D_DIM];
    // [plane][tile(0..7) * 512 + lane * 16 + j], 8 KB per plane
    __shared__ __align__(32) bf16_t sA[2][8 * 32 * 16];
    __shared__ __align__(32) bf16_t sB[2][8 * 32 * 16];

    const int tid = threadIdx.x;
    for (int d = tid; d < D_DIM; d += 256) {
        float sc = gamma[d] * rsqrtf(var[d] + BN_EPS);
        s_scale[d] = sc;
        s_shift[d] = beta[d] - mean[d] * sc;
    }

    const int lane   = tid & 31;
    const int waveId = tid >> 5;
    const int wm     = waveId & 3;          // 0..3 -> M offset within WG
    const int wn     = waveId >> 2;         // 0..1 -> N offset within WG
    const int blockM = blockIdx.x * 128;
    const int blockN = blockIdx.y * 128;
    const int lhalf  = lane & 31 & 15;      // row/col within 16
    const int khalf  = (lane >> 4) << 3;    // 0 (lanes 0-15) or 8 (lanes 16-31)

    // Staging assignment: thread t owns (tile = waveId, frag-lane = lane).
    const float* aRowP = priors + (size_t)(blockM + waveId * 16 + lhalf) * D_DIM;
    const float* aRowF = feat   + (size_t)(blockM + waveId * 16 + lhalf) * D_DIM;
    const float* bRowW = W      + (size_t)(blockN + waveId * 16 + lhalf) * D_DIM;
    bf16_t* sAhi = &sA[0][tid * 16];
    bf16_t* sAlo = &sA[1][tid * 16];
    bf16_t* sBhi = &sB[0][tid * 16];
    bf16_t* sBlo = &sB[1][tid * 16];

    v8f acc[2][4];
    #pragma unroll
    for (int mi = 0; mi < 2; ++mi)
        #pragma unroll
        for (int ni = 0; ni < 4; ++ni)
            acc[mi][ni] = (v8f){0.f,0.f,0.f,0.f,0.f,0.f,0.f,0.f};

    __syncthreads();   // s_scale/s_shift ready

    for (int k0 = 0; k0 < D_DIM; k0 += 32) {
        const int kb = k0 + khalf;

        // ---- cooperative stage: convert A/B block into split-bf16 LDS ----
        #pragma unroll
        for (int j = 0; j < 16; ++j) {
            const int k = kb + j + (j & 8);   // runs {kb..kb+7, kb+16..kb+23}

            float x = fmaf(aRowP[k] * aRowF[k], s_scale[k], s_shift[k]);
            bf16_t xh = (bf16_t)x;
            sAhi[j] = xh;
            sAlo[j] = (bf16_t)(x - (float)xh);

            float w = bRowW[k];
            bf16_t wh = (bf16_t)w;
            sBhi[j] = wh;
            sBlo[j] = (bf16_t)(w - (float)wh);
        }
        __syncthreads();   // staged data visible

        // ---- read frags + 24 WMMAs ----
        v16bf aHi[2], aLo[2];
        #pragma unroll
        for (int mi = 0; mi < 2; ++mi) {
            const int tA = wm * 2 + mi;
            aHi[mi] = *(const v16bf*)&sA[0][(tA * 32 + lane) * 16];
            aLo[mi] = *(const v16bf*)&sA[1][(tA * 32 + lane) * 16];
        }
        #pragma unroll
        for (int ni = 0; ni < 4; ++ni) {
            const int tB = wn * 4 + ni;
            v16bf bHi = *(const v16bf*)&sB[0][(tB * 32 + lane) * 16];
            v16bf bLo = *(const v16bf*)&sB[1][(tB * 32 + lane) * 16];
            #pragma unroll
            for (int mi = 0; mi < 2; ++mi) {
                v8f c = acc[mi][ni];
                c = __builtin_amdgcn_wmma_f32_16x16x32_bf16(
                        false, aHi[mi], false, bHi, (short)0, c, false, false);
                c = __builtin_amdgcn_wmma_f32_16x16x32_bf16(
                        false, aLo[mi], false, bHi, (short)0, c, false, false);
                c = __builtin_amdgcn_wmma_f32_16x16x32_bf16(
                        false, aHi[mi], false, bLo, (short)0, c, false, false);
                acc[mi][ni] = c;
            }
        }
        __syncthreads();   // protect LDS before next stage overwrites
    }

    // ---- epilogue: add bias, store C (16x16 f32: VGPR r -> M=r / M=r+8) ----
    const int baseM  = blockM + wm * 32;
    const int baseN  = blockN + wn * 64;
    const int rowOff = (lane >> 4) << 3;   // lanes 16-31 hold M = r + 8
    #pragma unroll
    for (int mi = 0; mi < 2; ++mi) {
        #pragma unroll
        for (int ni = 0; ni < 4; ++ni) {
            const int col = baseN + ni * 16 + lhalf;
            const float bs = bias[col];
            #pragma unroll
            for (int r = 0; r < 8; ++r) {
                const int row = baseM + mi * 16 + rowOff + r;
                out[(size_t)row * D_DIM + col] = acc[mi][ni][r] + bs;
            }
        }
    }
}

// ---------------------------------------------------------------------------
// Kernel 2: sparsemax over each 512-wide row, one wave32 per row.
// Michelot fixed-point iteration: tau <- (sum_{z>tau} z - 1)/|{z>tau}|,
// starting tau0=(sum-1)/512. Monotone and exact; 16 iters >> enough.
// In-place on d_out.
// ---------------------------------------------------------------------------
__global__ __launch_bounds__(256)
void at_sparsemax_kernel(float* __restrict__ y)
{
    const int lane = threadIdx.x & 31;
    const int wib  = threadIdx.x >> 5;
    const size_t row = (size_t)blockIdx.x * 8 + wib;
    float* p = y + row * D_DIM + lane * 16;

    float v[16];
    #pragma unroll
    for (int i = 0; i < 4; ++i) {
        float4 t = ((const float4*)p)[i];
        v[4*i+0] = t.x; v[4*i+1] = t.y; v[4*i+2] = t.z; v[4*i+3] = t.w;
    }

    float sm = 0.f;
    #pragma unroll
    for (int i = 0; i < 16; ++i) sm += v[i];
    #pragma unroll
    for (int o = 16; o >= 1; o >>= 1) sm += __shfl_xor(sm, o, 32);

    float tau = (sm - 1.0f) * (1.0f / (float)D_DIM);

    for (int it = 0; it < 16; ++it) {
        float s = 0.f, cnt = 0.f;
        #pragma unroll
        for (int i = 0; i < 16; ++i) {
            if (v[i] > tau) { s += v[i]; cnt += 1.0f; }
        }
        #pragma unroll
        for (int o = 16; o >= 1; o >>= 1) {
            s   += __shfl_xor(s,   o, 32);
            cnt += __shfl_xor(cnt, o, 32);
        }
        tau = (s - 1.0f) / cnt;   // argmax always in support -> cnt >= 1
    }

    #pragma unroll
    for (int i = 0; i < 4; ++i) {
        float4 t;
        t.x = fmaxf(v[4*i+0] - tau, 0.f);
        t.y = fmaxf(v[4*i+1] - tau, 0.f);
        t.z = fmaxf(v[4*i+2] - tau, 0.f);
        t.w = fmaxf(v[4*i+3] - tau, 0.f);
        ((float4*)p)[i] = t;
    }
}

// ---------------------------------------------------------------------------
extern "C" void kernel_launch(void* const* d_in, const int* in_sizes, int n_in,
                              void* d_out, int out_size, void* d_ws, size_t ws_size,
                              hipStream_t stream)
{
    const float* priors = (const float*)d_in[0];
    const float* feat   = (const float*)d_in[1];
    const float* gamma  = (const float*)d_in[2];
    const float* beta   = (const float*)d_in[3];
    const float* mean   = (const float*)d_in[4];
    const float* var    = (const float*)d_in[5];
    const float* fc_w   = (const float*)d_in[6];
    const float* fc_b   = (const float*)d_in[7];
    float* out = (float*)d_out;

    dim3 gGemm(BATCHN / 128, D_DIM / 128);   // 1024 x 4 workgroups
    at_gemm_bn_kernel<<<gGemm, 256, 0, stream>>>(
        priors, feat, gamma, beta, mean, var, fc_w, fc_b, out);

    at_sparsemax_kernel<<<BATCHN / 8, 256, 0, stream>>>(out);
}
